// GraphEdgeWeight_47493748359688
// MI455X (gfx1250) — matrix operand order repaired and checked by reading, not compiled, not run on hardware
//
#include <hip/hip_runtime.h>
#include <hip/hip_bf16.h>
#include <math.h>

typedef __attribute__((ext_vector_type(4)))  float     v4f;
typedef __attribute__((ext_vector_type(8)))  float     v8f;
typedef __attribute__((ext_vector_type(8)))  _Float16  v8h;
typedef __attribute__((ext_vector_type(16))) _Float16  v16h;

#define HDIM 128

union FragA {
  v16h         v;
  v8h          h[2];
  unsigned int u[8];
};

__device__ __forceinline__ v8h ld8_f32_to_f16(const float* p) {
  v4f a = *(const v4f*)p;
  v4f b = *(const v4f*)(p + 4);
  v8f t = __builtin_shufflevector(a, b, 0, 1, 2, 3, 4, 5, 6, 7);
  return __builtin_convertvector(t, v8h);
}

// B fragments pre-packed: fragment f, lane l owns 16 contiguous halves (32 B).
__device__ __forceinline__ v16h ldB(const _Float16* __restrict__ pk, int frag, int lane) {
  const _Float16* p = pk + (((size_t)frag * 32 + (size_t)lane) << 4);
  FragA f;
  f.h[0] = *(const v8h*)p;
  f.h[1] = *(const v8h*)(p + 8);
  return f.v;
}

// leaky_relu(v) = max(v, 0.01*v)  (valid because slope < 1)
__device__ __forceinline__ float leaky(float v) { return fmaxf(v, 0.01f * v); }

// Repack row-major f32 weight W[K][128] into f16 WMMA B-fragment layout.
// Fragment f = t*KC + c covers N-tile t (16 cols) x K-chunk c (32 rows).
// 16-bit B layout: lanes 0-15 hold K = 32c + {0..7,16..23}, lanes 16-31 hold
// K = 32c + {8..15,24..31}; column N = 16t + (lane&15).
__global__ void repack_w(const float* __restrict__ W, _Float16* __restrict__ dst,
                         int KC, int total) {
  int id = blockIdx.x * blockDim.x + threadIdx.x;
  if (id >= total) return;
  int j = id & 15;
  int l = (id >> 4) & 31;
  int f = id >> 9;
  int c = f % KC;
  int t = f / KC;
  int n = (t << 4) | (l & 15);
  int k = 32 * c + ((l >> 4) << 3) + (j & 7) + ((j >> 3) << 4);
  dst[id] = (_Float16)W[k * HDIM + n];
}

__global__ __launch_bounds__(128)
void edge_weight_kernel(const float* __restrict__ x,
                        const int* __restrict__ eidx,
                        const _Float16* __restrict__ Wp_pk, const float* __restrict__ bp,
                        const _Float16* __restrict__ Wc_pk, const float* __restrict__ bc,
                        const _Float16* __restrict__ W1_pk, const float* __restrict__ b1,
                        const float* __restrict__ W2, const float* __restrict__ b2,
                        float* __restrict__ out, int E) {
  // 4 waves per block, each wave owns a 16-edge tile and its own LDS region.
  // Row pitch 264 halves (528 B) -> conflict-free ds_load_b128.
  __shared__ __attribute__((aligned(16))) _Float16 lds[4][16][264];

  const int wave = threadIdx.x >> 5;
  const int lane = threadIdx.x & 31;
  const int l15  = lane & 15;
  const int hi   = lane >> 4;

  if ((long long)blockIdx.x * 64 >= (long long)E) return;

  const long long tile = (long long)blockIdx.x * 4 + wave;
  const long long e0 = tile * 16;

  long long eg = e0 + l15;
  if (eg > (long long)E - 1) eg = (long long)E - 1;
  const int src = eidx[eg];
  const int dst = eidx[(size_t)E + (size_t)eg];
  const float* prow = x + (size_t)src * HDIM;
  const float* crow = x + (size_t)dst * HDIM;

  // ---- A fragments: par & cld, 4 K-chunks of 32 (16-bit A layout:
  // lanes 0-15: K {0..7,16..23}+32c ; lanes 16-31: K {8..15,24..31}+32c) ----
  FragA ap[4], ac[4];
#pragma unroll
  for (int c = 0; c < 4; ++c) {
    const int k0 = 32 * c + hi * 8;
    ap[c].h[0] = ld8_f32_to_f16(prow + k0);
    ap[c].h[1] = ld8_f32_to_f16(prow + k0 + 16);
    ac[c].h[0] = ld8_f32_to_f16(crow + k0);
    ac[c].h[1] = ld8_f32_to_f16(crow + k0 + 16);
  }

  // ---- diff = |par - cld| in the same A layout:
  // packed f16 subtract, then clear sign bits (abs) with one AND per dword ----
  FragA ad[4];
#pragma unroll
  for (int c = 0; c < 4; ++c) {
    FragA d;
    d.v = ap[c].v - ac[c].v;
#pragma unroll
    for (int w = 0; w < 8; ++w) ad[c].u[w] = d.u[w] & 0x7fff7fffu;
  }

  // ---- GEMM1a: enc_p = leaky_relu(par @ W_p + b_p) -> LDS cols [0,128) ----
#pragma unroll
  for (int t = 0; t < 8; ++t) {
    const float bv = bp[16 * t + l15];
    v8f acc;
#pragma unroll
    for (int i = 0; i < 8; ++i) acc[i] = bv;
#pragma unroll
    for (int c = 0; c < 4; ++c) {
      v16h bf = ldB(Wp_pk, t * 4 + c, lane);
      acc = __builtin_amdgcn_wmma_f32_16x16x32_f16(false, ap[c].v, false, bf,
                                                   (short)0, acc, false, false);
    }
#pragma unroll
    for (int r = 0; r < 8; ++r) {
      // C layout: M = r + 8*hi, N = 16t + l15
      lds[wave][r + 8 * hi][16 * t + l15] = (_Float16)leaky(acc[r]);
    }
  }

  // ---- GEMM1b: enc_c = leaky_relu(cld @ W_c + b_c) -> LDS cols [128,256) ----
#pragma unroll
  for (int t = 0; t < 8; ++t) {
    const float bv = bc[16 * t + l15];
    v8f acc;
#pragma unroll
    for (int i = 0; i < 8; ++i) acc[i] = bv;
#pragma unroll
    for (int c = 0; c < 4; ++c) {
      v16h bf = ldB(Wc_pk, t * 4 + c, lane);
      acc = __builtin_amdgcn_wmma_f32_16x16x32_f16(false, ac[c].v, false, bf,
                                                   (short)0, acc, false, false);
    }
#pragma unroll
    for (int r = 0; r < 8; ++r) {
      lds[wave][r + 8 * hi][128 + 16 * t + l15] = (_Float16)leaky(acc[r]);
    }
  }

  __syncthreads();

  // ---- GEMM2: h = relu([enc_p|enc_c|diff] @ W1 + b1), K = 384 ----
  v8f h8[8];
#pragma unroll
  for (int t = 0; t < 8; ++t) {
    const float bv = b1[16 * t + l15];
#pragma unroll
    for (int i = 0; i < 8; ++i) h8[t][i] = bv;
  }

  const _Float16* lrow = &lds[wave][l15][0];
#pragma unroll
  for (int c2 = 0; c2 < 12; ++c2) {
    FragA a2;
    if (c2 < 8) {                       // enc_p / enc_c from LDS
      const int k0 = 32 * c2 + hi * 8;
      a2.h[0] = *(const v8h*)(lrow + k0);
      a2.h[1] = *(const v8h*)(lrow + k0 + 16);
    } else {                            // diff straight from registers
      a2 = ad[c2 - 8];
    }
#pragma unroll
    for (int t = 0; t < 8; ++t) {
      v16h bf = ldB(W1_pk, t * 12 + c2, lane);
      h8[t] = __builtin_amdgcn_wmma_f32_16x16x32_f16(false, a2.v, false, bf,
                                                     (short)0, h8[t], false, false);
    }
  }

  // ---- Final: out = sigmoid(relu(h) @ W2 + b2) ----
  float partial[8];
#pragma unroll
  for (int r = 0; r < 8; ++r) partial[r] = 0.0f;
#pragma unroll
  for (int t = 0; t < 8; ++t) {
    const float w = W2[16 * t + l15];
#pragma unroll
    for (int r = 0; r < 8; ++r) {
      partial[r] += fmaxf(h8[t][r], 0.0f) * w;
    }
  }
#pragma unroll
  for (int r = 0; r < 8; ++r) {   // reduce within each 16-lane half
    partial[r] += __shfl_xor(partial[r], 8, 32);
    partial[r] += __shfl_xor(partial[r], 4, 32);
    partial[r] += __shfl_xor(partial[r], 2, 32);
    partial[r] += __shfl_xor(partial[r], 1, 32);
  }

  const float b2v = b2[0];
  if (l15 == 0) {                 // lanes 0 & 16 write edges [e0..e0+7] / [e0+8..e0+15]
#pragma unroll
    for (int r = 0; r < 8; ++r) {
      long long e = e0 + 8 * hi + r;
      if (e < (long long)E) {
        float v = partial[r] + b2v;
        out[e] = 1.0f / (1.0f + expf(-v));
      }
    }
  }
}

extern "C" void kernel_launch(void* const* d_in, const int* in_sizes, int n_in,
                              void* d_out, int out_size, void* d_ws, size_t ws_size,
                              hipStream_t stream) {
  const float* x  = (const float*)d_in[0];
  const int*   ei = (const int*)d_in[1];
  const float* Wp = (const float*)d_in[2];
  const float* bp = (const float*)d_in[3];
  const float* Wc = (const float*)d_in[4];
  const float* bc = (const float*)d_in[5];
  const float* W1 = (const float*)d_in[6];
  const float* b1 = (const float*)d_in[7];
  const float* W2 = (const float*)d_in[8];
  const float* b2 = (const float*)d_in[9];
  float* out = (float*)d_out;
  const int E = in_sizes[1] / 2;

  _Float16* wsh   = (_Float16*)d_ws;
  _Float16* Wp_pk = wsh;                  // 8*4*32*16  = 16384 halves (32 KB)
  _Float16* Wc_pk = wsh + 16384;
  _Float16* W1_pk = wsh + 32768;          // 8*12*32*16 = 49152 halves (96 KB)

  repack_w<<<(16384 + 255) / 256, 256, 0, stream>>>(Wp, Wp_pk, 4, 16384);
  repack_w<<<(16384 + 255) / 256, 256, 0, stream>>>(Wc, Wc_pk, 4, 16384);
  repack_w<<<(49152 + 255) / 256, 256, 0, stream>>>(W1, W1_pk, 12, 49152);

  const long long tiles  = ((long long)E + 15) / 16;
  const long long blocks = (tiles + 3) / 4;
  edge_weight_kernel<<<dim3((unsigned)blocks), dim3(128), 0, stream>>>(
      x, ei, Wp_pk, bp, Wc_pk, bc, W1_pk, b1, W2, b2, out, E);
  (void)n_in; (void)out_size; (void)ws_size;
}
